// MSLA_60000693125698
// MI455X (gfx1250) — compile-verified, matrix operand order checked by reading, not run on hardware
//
#include <hip/hip_runtime.h>
#include <hip/hip_bf16.h>

// MSLA for MI455X (gfx1250): bf16 WMMA GEMMs with TDM (tensor_load_to_lds)
// double-buffered tile staging + fused latent-attention.
// K-projection (Wk/bk) is dead code in the reference -> skipped.

typedef __bf16 bf16_t;
typedef __attribute__((ext_vector_type(16))) __bf16 v16bf;
typedef __attribute__((ext_vector_type(8)))  float  v8f;
typedef __attribute__((ext_vector_type(4)))  unsigned int v4u;
typedef __attribute__((ext_vector_type(8)))  unsigned int v8u;

#define MDIM 16384   // B*T
#define DDIM 2048    // D
#define HHEADS 16
#define HDIM 128     // D / H
#define LLAT 64
#define KTOP 4

union FragB { uint4 u[2]; v16bf v; };

// ---------------------------------------------------------------------------
// TDM: DMA a (tile_rows x 32) bf16 tile, row stride = stride_elems, into LDS
// with 16B padding after every 64B row (-> LDS row stride 40 bf16 / 80B).
// D# per CDNA5 ISA ch.8: group0 = count/lds_addr/global_addr/type,
// group1 = data_size|pad | tensor dims | tile dims | dim0 stride.
// ---------------------------------------------------------------------------
__device__ __forceinline__ void tdm_load_2d(unsigned int lds_byte_off,
                                            const void* gsrc,
                                            unsigned int stride_elems,
                                            unsigned int tile_rows) {
  unsigned long long ga = (unsigned long long)(uintptr_t)gsrc;
  v4u g0;
  g0[0] = 1u;                                            // count=1 (valid D#)
  g0[1] = lds_byte_off;                                  // lds_addr
  g0[2] = (unsigned int)ga;                              // global_addr[31:0]
  g0[3] = (unsigned int)((ga >> 32) & 0x01FFFFFFu) | (2u << 30); // addr[56:32] | type=2
  v8u g1;
  g1[0] = (1u << 16)        // data_size = 1 -> 2 bytes
        | (1u << 20)        // pad_enable
        | (3u << 22)        // pad_interval: 16 DWORDs (64B = one 32-elem row)
        | (3u << 25);       // pad_amount:   4 DWORDs (16B = 8 bf16)
  const unsigned int td0 = 1u << 20;                     // tensor_dim0 (OOB bound, >> tile)
  const unsigned int td1 = 1u << 20;                     // tensor_dim1
  g1[1] = (td0 & 0xFFFFu) << 16;                         // [63:48] tensor_dim0 lo
  g1[2] = (td0 >> 16) | ((td1 & 0xFFFFu) << 16);         // dim0 hi | dim1 lo
  g1[3] = (td1 >> 16) | (32u << 16);                     // dim1 hi | tile_dim0 = 32
  g1[4] = tile_rows;                                     // tile_dim1 ; tile_dim2 = 0 (2D)
  g1[5] = stride_elems;                                  // tensor_dim0_stride[31:0]
  g1[6] = 0u;                                            // stride hi | dim1_stride lo
  g1[7] = 0u;
  asm volatile("tensor_load_to_lds %0, %1" :: "s"(g0), "s"(g1) : "memory");
}

// ---------------------------------------------------------------------------
// f32 -> bf16 bulk convert (4 elems/thread, 8-byte packed stores)
// ---------------------------------------------------------------------------
__global__ __launch_bounds__(256) void msla_cvt_bf16(const float* __restrict__ s,
                                                     bf16_t* __restrict__ d, int n4) {
  int i = blockIdx.x * blockDim.x + threadIdx.x;
  if (i >= n4) return;
  float4 v = reinterpret_cast<const float4*>(s)[i];
  union { bf16_t b[4]; unsigned long long u; } pk;
  pk.b[0] = (bf16_t)v.x; pk.b[1] = (bf16_t)v.y;
  pk.b[2] = (bf16_t)v.z; pk.b[3] = (bf16_t)v.w;
  reinterpret_cast<unsigned long long*>(d)[i] = pk.u;
}

// latent (L, D) f32 -> per-head (H, L, HD) bf16
__global__ __launch_bounds__(256) void msla_reorder_latent(const float* __restrict__ lat,
                                                           bf16_t* __restrict__ out) {
  int i = blockIdx.x * blockDim.x + threadIdx.x;
  if (i >= LLAT * DDIM) return;
  int l = i >> 11;
  int rem = i & 2047;
  int h = rem >> 7;
  int hd = rem & 127;
  out[((size_t)h * LLAT + l) * HDIM + hd] = (bf16_t)lat[i];
}

// ---------------------------------------------------------------------------
// WMMA GEMM: C[m,n] = sum_k X[m,k] * W[n,k] + bias[n]
// Block tile 128x128, BK=32, 8 waves each computing 32(M) x 64(N).
// Staging: per wave, two TDM tensor_load_to_lds ops (16-row A slice + 16-row
// B slice) per K-slab, double-buffered; s_wait_tensorcnt + barrier to sync.
// MODE 0: fused Q|V projection (N=4096): n<2048 -> Q (bf16), else V (f32)
// MODE 1: output projection   (N=2048): f32 out
// ---------------------------------------------------------------------------
template <int MODE>
__global__ __launch_bounds__(256) void msla_gemm_wmma(
    const bf16_t* __restrict__ X,
    const bf16_t* __restrict__ W0, const bf16_t* __restrict__ W1,
    const float* __restrict__ b0, const float* __restrict__ b1,
    bf16_t* __restrict__ outQ, float* __restrict__ outV, float* __restrict__ outF) {
  constexpr int LDT = 40;                       // padded LDS row stride (bf16)
  __shared__ bf16_t As[2][128 * LDT];
  __shared__ bf16_t Bs[2][128 * LDT];

  const int tid  = threadIdx.x;
  const int lane = tid & 31;
  const int wv   = tid >> 5;
  const int wm   = (wv & 3) * 32;               // wave M offset within tile
  const int wn   = (wv >> 2) * 64;              // wave N offset within tile
  const int r16  = lane & 15;
  const int hi8  = (lane >> 4) * 8;             // K-chunk shift for lanes 16..31

  const int bm = blockIdx.y * 128;
  const int bn = blockIdx.x * 128;

  // Uniform weight base for this block's 128 N-rows (bn is 128-aligned, so a
  // block never straddles the Wq/Wv boundary at n=2048).
  const bf16_t* wbase;
  if (MODE == 0) wbase = (bn < DDIM) ? (W0 + (size_t)bn * DDIM)
                                     : (W1 + (size_t)(bn - DDIM) * DDIM);
  else           wbase = W0 + (size_t)bn * DDIM;

  const unsigned int asOff  = (unsigned int)(uintptr_t)&As[0][0];
  const unsigned int bsOff  = (unsigned int)(uintptr_t)&Bs[0][0];
  const unsigned int bufB   = 128 * LDT * 2;    // bytes per buffer
  const unsigned int sliceB = (unsigned int)(wv * 16 * LDT * 2);
  const bf16_t* aSrc = X + (size_t)(bm + wv * 16) * DDIM;
  const bf16_t* bSrc = wbase + (size_t)(wv * 16) * DDIM;

  v8f acc[2][4];
  #pragma unroll
  for (int im = 0; im < 2; ++im)
    #pragma unroll
    for (int in = 0; in < 4; ++in) acc[im][in] = {};

  // prologue: DMA first K-slab into buffer 0
  tdm_load_2d(asOff + sliceB, aSrc, DDIM, 16);
  tdm_load_2d(bsOff + sliceB, bSrc, DDIM, 16);

  constexpr int NK = DDIM / 32;
  for (int i = 0; i < NK; ++i) {
    const int buf = i & 1;
    if (i + 1 < NK) {                           // DMA next K-slab into other buffer
      const unsigned int bo = (unsigned int)((buf ^ 1) * bufB);
      tdm_load_2d(asOff + bo + sliceB, aSrc + (i + 1) * 32, DDIM, 16);
      tdm_load_2d(bsOff + bo + sliceB, bSrc + (i + 1) * 32, DDIM, 16);
      __builtin_amdgcn_s_wait_tensorcnt(2);     // TDM in-order: slab i complete
    } else {
      __builtin_amdgcn_s_wait_tensorcnt(0);
    }
    __syncthreads();                            // all waves' slices of slab i ready

    FragB a[2], b[4];
    #pragma unroll
    for (int im = 0; im < 2; ++im) {
      const bf16_t* p = &As[buf][(wm + im * 16 + r16) * LDT + hi8];
      a[im].u[0] = *reinterpret_cast<const uint4*>(p);
      a[im].u[1] = *reinterpret_cast<const uint4*>(p + 16);
    }
    #pragma unroll
    for (int in = 0; in < 4; ++in) {
      const bf16_t* p = &Bs[buf][(wn + in * 16 + r16) * LDT + hi8];
      b[in].u[0] = *reinterpret_cast<const uint4*>(p);
      b[in].u[1] = *reinterpret_cast<const uint4*>(p + 16);
    }
    #pragma unroll
    for (int im = 0; im < 2; ++im)
      #pragma unroll
      for (int in = 0; in < 4; ++in)
        acc[im][in] = __builtin_amdgcn_wmma_f32_16x16x32_bf16(
            false, a[im].v, false, b[in].v, (short)0, acc[im][in], false, false);
    __syncthreads();                            // reads done before buf is re-filled
  }

  // epilogue: C row = e + hi8 (per ISA C/D layout), col = r16
  #pragma unroll
  for (int im = 0; im < 2; ++im) {
    #pragma unroll
    for (int in = 0; in < 4; ++in) {
      #pragma unroll
      for (int e = 0; e < 8; ++e) {
        int m = bm + wm + im * 16 + e + hi8;
        int n = bn + wn + in * 16 + r16;
        float c = acc[im][in][e];
        if (MODE == 0) {
          if (n < DDIM) outQ[(size_t)m * DDIM + n] = (bf16_t)(c + b0[n]);
          else          outV[(size_t)m * DDIM + (n - DDIM)] = c + b1[n - DDIM];
        } else {
          outF[(size_t)m * DDIM + n] = c + b0[n];
        }
      }
    }
  }
}

// ---------------------------------------------------------------------------
// Latent attention: per 16-row tile, each wave handles 2 heads.
// Logits 16x64 via WMMA (Q 16x128 x latent^T 128x64), top-4 softmax,
// rank-4 weighted latent sum + V, bf16 output for final GEMM.
// ---------------------------------------------------------------------------
__global__ __launch_bounds__(256) void msla_attn(
    const bf16_t* __restrict__ Qbf, const bf16_t* __restrict__ latB,
    const float* __restrict__ latF, const float* __restrict__ Vf,
    bf16_t* __restrict__ Abf) {
  __shared__ float lg[8][16][64];
  const int tid  = threadIdx.x;
  const int lane = tid & 31;
  const int wv   = tid >> 5;
  const int r16  = lane & 15;
  const int hi8  = (lane >> 4) * 8;
  const int m0   = blockIdx.x * 16;
  const float scale = 0.08838834764831845f;     // 1/sqrt(128)

  for (int hh = 0; hh < 2; ++hh) {
    const int h = wv + hh * 8;

    v8f acc[4];
    #pragma unroll
    for (int in = 0; in < 4; ++in) acc[in] = {};

    #pragma unroll
    for (int kk = 0; kk < 4; ++kk) {
      FragB a;
      const bf16_t* ap = Qbf + (size_t)(m0 + r16) * DDIM + h * HDIM + kk * 32 + hi8;
      a.u[0] = *reinterpret_cast<const uint4*>(ap);
      a.u[1] = *reinterpret_cast<const uint4*>(ap + 16);
      #pragma unroll
      for (int in = 0; in < 4; ++in) {
        FragB b;
        const bf16_t* bp = latB + ((size_t)h * LLAT + in * 16 + r16) * HDIM + kk * 32 + hi8;
        b.u[0] = *reinterpret_cast<const uint4*>(bp);
        b.u[1] = *reinterpret_cast<const uint4*>(bp + 16);
        acc[in] = __builtin_amdgcn_wmma_f32_16x16x32_bf16(
            false, a.v, false, b.v, (short)0, acc[in], false, false);
      }
    }
    #pragma unroll
    for (int in = 0; in < 4; ++in)
      #pragma unroll
      for (int e = 0; e < 8; ++e)
        lg[wv][e + hi8][in * 16 + r16] = acc[in][e] * scale;
    __syncthreads();

    {
      const int row = r16;
      const int m = m0 + row;
      const float* lr = lg[wv][row];
      float bv4[KTOP]; int bi4[KTOP];
      #pragma unroll
      for (int j = 0; j < KTOP; ++j) { bv4[j] = -1e30f; bi4[j] = 0; }
      for (int l = 0; l < LLAT; ++l) {
        float x = lr[l];
        if (x > bv4[KTOP - 1]) {
          int j = KTOP - 1;
          while (j > 0 && x > bv4[j - 1]) { bv4[j] = bv4[j - 1]; bi4[j] = bi4[j - 1]; --j; }
          bv4[j] = x; bi4[j] = l;
        }
      }
      float ex[KTOP], sum = 0.f;
      #pragma unroll
      for (int j = 0; j < KTOP; ++j) { ex[j] = __expf(bv4[j] - bv4[0]); sum += ex[j]; }
      const float inv = 1.f / sum;
      float w4[KTOP];
      #pragma unroll
      for (int j = 0; j < KTOP; ++j) w4[j] = ex[j] * inv;

      const float* l0 = latF + (size_t)bi4[0] * DDIM + h * HDIM;
      const float* l1 = latF + (size_t)bi4[1] * DDIM + h * HDIM;
      const float* l2 = latF + (size_t)bi4[2] * DDIM + h * HDIM;
      const float* l3 = latF + (size_t)bi4[3] * DDIM + h * HDIM;
      const int hd0 = (lane >> 4) * 64;
      const size_t obase = (size_t)m * DDIM + h * HDIM;
      for (int d = hd0; d < hd0 + 64; ++d) {
        float val = w4[0] * l0[d] + w4[1] * l1[d] + w4[2] * l2[d] + w4[3] * l3[d];
        val += Vf[obase + d];
        Abf[obase + d] = (bf16_t)val;
      }
    }
    __syncthreads();
  }
}

// ---------------------------------------------------------------------------
extern "C" void kernel_launch(void* const* d_in, const int* in_sizes, int n_in,
                              void* d_out, int out_size, void* d_ws, size_t ws_size,
                              hipStream_t stream) {
  const float* hidden = (const float*)d_in[0];
  const float* Wq = (const float*)d_in[1];
  const float* bq = (const float*)d_in[2];
  // d_in[3]=Wk, d_in[4]=bk: K-projection is unused by the reference output.
  const float* Wv = (const float*)d_in[5];
  const float* bv = (const float*)d_in[6];
  const float* Wo = (const float*)d_in[7];
  const float* bo = (const float*)d_in[8];
  const float* lat = (const float*)d_in[9];
  (void)in_sizes; (void)n_in; (void)out_size; (void)ws_size;

  char* ws = (char*)d_ws;
  bf16_t* Xbf  = (bf16_t*)ws; ws += (size_t)MDIM * DDIM * sizeof(bf16_t);
  bf16_t* WqB  = (bf16_t*)ws; ws += (size_t)DDIM * DDIM * sizeof(bf16_t);
  bf16_t* WvB  = (bf16_t*)ws; ws += (size_t)DDIM * DDIM * sizeof(bf16_t);
  bf16_t* WoB  = (bf16_t*)ws; ws += (size_t)DDIM * DDIM * sizeof(bf16_t);
  bf16_t* latB = (bf16_t*)ws; ws += (size_t)LLAT * DDIM * sizeof(bf16_t);
  bf16_t* Qbf  = (bf16_t*)ws; ws += (size_t)MDIM * DDIM * sizeof(bf16_t);
  float*  Vf   = (float*)ws;  ws += (size_t)MDIM * DDIM * sizeof(float);
  bf16_t* Abf  = (bf16_t*)ws; ws += (size_t)MDIM * DDIM * sizeof(bf16_t);

  const int XB4 = MDIM * DDIM / 4;
  const int WB4 = DDIM * DDIM / 4;
  msla_cvt_bf16<<<(XB4 + 255) / 256, 256, 0, stream>>>(hidden, Xbf, XB4);
  msla_cvt_bf16<<<(WB4 + 255) / 256, 256, 0, stream>>>(Wq, WqB, WB4);
  msla_cvt_bf16<<<(WB4 + 255) / 256, 256, 0, stream>>>(Wv, WvB, WB4);
  msla_cvt_bf16<<<(WB4 + 255) / 256, 256, 0, stream>>>(Wo, WoB, WB4);
  msla_reorder_latent<<<(LLAT * DDIM + 255) / 256, 256, 0, stream>>>(lat, latB);

  // fused Q|V projection: N = 4096
  msla_gemm_wmma<0><<<dim3(4096 / 128, MDIM / 128), 256, 0, stream>>>(
      Xbf, WqB, WvB, bq, bv, Qbf, Vf, nullptr);

  // latent attention
  msla_attn<<<MDIM / 16, 256, 0, stream>>>(Qbf, latB, lat, Vf, Abf);

  // output projection: N = 2048, f32 out
  msla_gemm_wmma<1><<<dim3(DDIM / 128, MDIM / 128), 256, 0, stream>>>(
      Abf, WoB, nullptr, bo, nullptr, nullptr, nullptr, (float*)d_out);
}
